// SmoothingLoss_60198261621420
// MI455X (gfx1250) — matrix operand order repaired and checked by reading, not compiled, not run on hardware
//
#include <hip/hip_runtime.h>
#include <hip/hip_bf16.h>

typedef __attribute__((ext_vector_type(16))) _Float16 v16h;
typedef __attribute__((ext_vector_type(8)))  float    v8f;

#define BATCH   8
#define CHN     3
#define H       512
#define W       512
#define RAD     10
#define TILE    16
#define SROWS   36            // TILE + 2*RAD
#define SSTR    564           // logical width 560 (cols -10..549), %4==0, bank-rotating
#define RSTR    524           // logical width 512, %4==0, bank-rotating
#define THREADS 256
#define NWG     (BATCH * (H / TILE))   // 256 workgroups

#define WMMA_F16(A, B, C) \
    __builtin_amdgcn_wmma_f32_16x16x32_f16(false, (A), false, (B), (short)0, (C), false, false)

__global__ __launch_bounds__(THREADS)
void smoothing_loss_main(const float* __restrict__ x,
                         const float* __restrict__ y,
                         const float* __restrict__ mask,
                         float* __restrict__ ws)
{
    __shared__ float s_tile[SROWS * SSTR];   // channel-reduced diff, zero-padded halo
    __shared__ float r_tile[SROWS * RSTR];   // horizontal 21-tap box sums
    __shared__ float red_num[THREADS];
    __shared__ float red_den[THREADS];

    const int tid  = threadIdx.x;
    const int wgid = blockIdx.x;
    const int b    = wgid >> 5;          // image
    const int h0   = (wgid & 31) << 4;   // output row base of this band
    const int wave = tid >> 5;
    const int lane = tid & 31;
    const int half = lane >> 4;          // lane half (WMMA K/M split)
    const int mrow = lane & 15;          // A-matrix M row / B-matrix N col for this lane

    // ---- Band matrices (constant, exact in f16).
    // A k order used here (contiguous, B matches; not the ISA table order): lanes<16 hold K 0-15, lanes>=16 hold K 16-31.
    // A0[m][k] = 1 iff m <= k <= m+20 (window chunk K=0..31)
    // A1[m][k] = 1 iff m <= k+32 <= m+20  ->  k <= m-12 (window chunk K=32..63)
    v16h A0, A1;
    #pragma unroll
    for (int i = 0; i < 16; ++i) {
        int k = half * 16 + i;
        A0[i] = (_Float16)((k >= mrow && k <= mrow + 20) ? 1.0f : 0.0f);
        A1[i] = (_Float16)(((k + 32) <= mrow + 20) ? 1.0f : 0.0f);
    }

    // ---- Stage 0: zero LDS (halo padding relies on this)
    for (int u = tid; u < SROWS * SSTR; u += THREADS) s_tile[u] = 0.0f;
    __syncthreads();

    // ---- Stage 1: s = sum_c (x - y), rows h0-10 .. h0+25, cols shifted +10
    {
        const int cs4 = (H * W) / 4;                // channel stride in float4
        for (int u = tid; u < SROWS * (W / 4); u += THREADS) {
            const int tr = u >> 7;                  // tile row 0..35
            const int c4 = u & 127;                 // float4 column block
            const int ar = h0 - RAD + tr;           // absolute image row
            if ((unsigned)ar < (unsigned)H) {
                const size_t p = ((size_t)(b * CHN) * H + ar) * W + (c4 << 2);
                const float4* xp = (const float4*)(x + p);
                const float4* yp = (const float4*)(y + p);
                __builtin_prefetch(xp + 16, 0, 1);  // global_prefetch_b8
                float4 xa = xp[0], xb = xp[cs4], xc = xp[2 * cs4];
                float4 ya = yp[0], yb = yp[cs4], yc = yp[2 * cs4];
                float* d = &s_tile[tr * SSTR + RAD + (c4 << 2)];
                d[0] = (xa.x - ya.x) + (xb.x - yb.x) + (xc.x - yc.x);
                d[1] = (xa.y - ya.y) + (xb.y - yb.y) + (xc.y - yc.y);
                d[2] = (xa.z - ya.z) + (xb.z - yb.z) + (xc.z - yc.z);
                d[3] = (xa.w - ya.w) + (xb.w - yb.w) + (xc.w - yc.w);
            }
        }
    }
    __syncthreads();

    // ---- Stage 2: horizontal 21-tap box sum via banded WMMA.
    // D[m][n] = sum_k band[m][k] * s[row rbase+n][w0 + k]  ->  r at (rbase+n, w0+m)
    for (int t = wave; t < 96; t += 8) {
        const int w0    = (t & 31) << 4;            // output col base
        const int rbase = (t >> 5) << 4;            // row group base: 0,16,32
        int brow = rbase + mrow;                    // B-matrix data row (clamped: band rows 36..47 discarded)
        if (brow > SROWS - 1) brow = SROWS - 1;
        const float* srow = &s_tile[brow * SSTR];

        v16h B0, B1;
        const int cb0 = w0 + half * 16;             // K chunk 0 (abs cols w0-10 .. )
        const int cb1 = w0 + 32 + half * 16;        // K chunk 1
        #pragma unroll
        for (int i = 0; i < 16; ++i) {
            B0[i] = (_Float16)srow[cb0 + i];
            B1[i] = (_Float16)srow[cb1 + i];
        }
        v8f acc = {};
        acc = WMMA_F16(A0, B0, acc);
        acc = WMMA_F16(A1, B1, acc);

        const int orow = rbase + mrow;              // D: lane&15 = N = row offset
        if (orow < SROWS) {
            float* dst = &r_tile[orow * RSTR + w0 + half * 8];   // v -> M = half*8+v (consec cols)
            float4 lo, hi;
            lo.x = acc[0]; lo.y = acc[1]; lo.z = acc[2]; lo.w = acc[3];
            hi.x = acc[4]; hi.y = acc[5]; hi.z = acc[6]; hi.w = acc[7];
            ((float4*)dst)[0] = lo;
            ((float4*)dst)[1] = hi;
        }
    }
    __syncthreads();

    // ---- Stage 3: vertical 21-tap box sum via banded WMMA + fused loss terms.
    float lnum = 0.0f, lden = 0.0f;
    for (int t = wave; t < 32; t += 8) {
        const int c0  = t << 4;                     // output col base
        const int col = c0 + mrow;                  // B: lane&15 = N = column
        v16h B0, B1;
        #pragma unroll
        for (int i = 0; i < 16; ++i) {
            const int r0 = half * 16 + i;           // rows 0..31
            int r1 = 32 + half * 16 + i;            // rows 32..35 live; rest band-zero
            if (r1 > SROWS - 1) r1 = SROWS - 1;     // clamp: finite data * zero band
            B0[i] = (_Float16)r_tile[r0 * RSTR + col];
            B1[i] = (_Float16)r_tile[r1 * RSTR + col];
        }
        v8f acc = {};
        acc = WMMA_F16(A0, B0, acc);
        acc = WMMA_F16(A1, B1, acc);

        // D[m][n] = 21x21 box sum at (h0+m, c0+n); conv = 441*s_center - box
        #pragma unroll
        for (int v = 0; v < 8; ++v) {
            const int m    = half * 8 + v;          // out row offset
            const int n    = mrow;                  // out col offset
            const int hrow = h0 + m;
            const int wcol = c0 + n;
            const float sc   = s_tile[(RAD + m) * SSTR + (RAD + wcol)];
            const float conv = 441.0f * sc - acc[v];
            const float mv   = mask[((size_t)b * H + hrow) * W + wcol];
            lnum += fabsf(conv) * mv;
            lden += mv;
        }
    }

    // ---- Stage 4: deterministic block reduction -> per-WG partials (no float atomics)
    red_num[tid] = lnum;
    red_den[tid] = lden;
    __syncthreads();
    #pragma unroll
    for (int s = THREADS / 2; s > 0; s >>= 1) {
        if (tid < s) {
            red_num[tid] += red_num[tid + s];
            red_den[tid] += red_den[tid + s];
        }
        __syncthreads();
    }
    if (tid == 0) {
        ws[wgid]       = red_num[0];
        ws[NWG + wgid] = red_den[0];
    }
}

__global__ __launch_bounds__(THREADS)
void smoothing_loss_finalize(const float* __restrict__ ws, float* __restrict__ out)
{
    __shared__ float rn[THREADS];
    __shared__ float rd[THREADS];
    const int tid = threadIdx.x;
    rn[tid] = ws[tid];
    rd[tid] = ws[NWG + tid];
    __syncthreads();
    #pragma unroll
    for (int s = THREADS / 2; s > 0; s >>= 1) {
        if (tid < s) { rn[tid] += rn[tid + s]; rd[tid] += rd[tid + s]; }
        __syncthreads();
    }
    if (tid == 0) out[0] = rn[0] / rd[0];
}

extern "C" void kernel_launch(void* const* d_in, const int* in_sizes, int n_in,
                              void* d_out, int out_size, void* d_ws, size_t ws_size,
                              hipStream_t stream)
{
    const float* x    = (const float*)d_in[0];   // (8,3,512,512) f32
    const float* y    = (const float*)d_in[1];   // (8,3,512,512) f32
    const float* mask = (const float*)d_in[2];   // (8,1,512,512) f32
    float* ws  = (float*)d_ws;                   // 2*NWG floats of partials
    float* out = (float*)d_out;                  // scalar loss

    smoothing_loss_main<<<NWG, THREADS, 0, stream>>>(x, y, mask, ws);
    smoothing_loss_finalize<<<1, THREADS, 0, stream>>>(ws, out);
}
